// UncertaintyAwareCausalLayer_68693706932589
// MI455X (gfx1250) — matrix operand order repaired and checked by reading, not compiled, not run on hardware
//
#include <hip/hip_runtime.h>
#include <hip/hip_fp16.h>

#define N_NODES 40000
#define N_EDGES 640000
#define D       128
#define LN_EPS  1e-5f

typedef __attribute__((ext_vector_type(16))) _Float16 v16h;
typedef __attribute__((ext_vector_type(8)))  float    v8f;

// ---------------------------------------------------------------------------
// Zero d_out (segment-sum accumulators) and degree each call.
// ---------------------------------------------------------------------------
__global__ void __launch_bounds__(256) zero_kernel(float* __restrict__ out,
                                                   float* __restrict__ degree) {
    const size_t total_out = (size_t)2 * N_NODES * D;
    size_t idx = (size_t)blockIdx.x * blockDim.x + threadIdx.x;
    if (idx < total_out) {
        out[idx] = 0.0f;
    } else if (idx < total_out + N_NODES) {
        degree[idx - total_out] = 0.0f;
    }
}

// ---------------------------------------------------------------------------
// Convert weights to f16 directly in WMMA B-fragment order:
//   frag t = (ntile*4 + kk)*32 + lane  holds 16 contiguous halves,
//   element j -> W[kk*32 + (lane>=16)*16 + j][ntile*16 + (lane&15)]
// Each lane's GEMM B-fragment load becomes one aligned 32-byte vector load.
// ---------------------------------------------------------------------------
__global__ void __launch_bounds__(256) convw_frag_kernel(const float* __restrict__ Wm,
                                                         const float* __restrict__ Wl,
                                                         _Float16* __restrict__ Wm16,
                                                         _Float16* __restrict__ Wl16) {
    const int t = blockIdx.x * blockDim.x + threadIdx.x;   // 0 .. 1023
    if (t >= 8 * 4 * 32) return;
    const int lane = t & 31;
    const int kk   = (t >> 5) & 3;
    const int nt   = t >> 7;
    const int ncol = nt * 16 + (lane & 15);
    const int kb   = kk * 32 + (lane >> 4) * 16;

    _Float16* __restrict__ dm = Wm16 + (size_t)t * 16;
    _Float16* __restrict__ dl = Wl16 + (size_t)t * 16;
    #pragma unroll
    for (int j = 0; j < 16; ++j) {
        dm[j] = (_Float16)Wm[(size_t)(kb + j) * D + ncol];
        dl[j] = (_Float16)Wl[(size_t)(kb + j) * D + ncol];
    }
}

// ---------------------------------------------------------------------------
// WMMA GEMM: h_mean = x@W_mean + b_mean ; h_var = exp(x@W_logvar + b_logvar)
// Block = 256 threads = 8 waves; wave w handles N-tile w; blockIdx.x = M-tile.
// Per wave: 4 k-steps x 2 matrices = 8 x v_wmma_f32_16x16x32_f16.
// A fragments: global_load_b128 (fp32 x) + v_cvt_pk_f16_f32.
// B fragments: single 32-byte vector load from pre-fragmented f16 weights.
// ---------------------------------------------------------------------------
__global__ void __launch_bounds__(256) gemm_kernel(const float*    __restrict__ x,
                                                   const _Float16* __restrict__ Wm,
                                                   const _Float16* __restrict__ Wl,
                                                   const float*    __restrict__ bm,
                                                   const float*    __restrict__ bl,
                                                   float* __restrict__ h_mean,
                                                   float* __restrict__ h_var) {
    const int wave  = threadIdx.x >> 5;      // 0..7 -> N tile
    const int lane  = threadIdx.x & 31;
    const int mtile = blockIdx.x;            // 0..2499
    const int l16   = lane & 15;
    const int hi    = lane >> 4;             // 0 or 1 (lane group)

    const int mrow = mtile * 16 + l16;       // A-fragment row for this lane
    const int ncol = wave * 16 + l16;        // B-fragment / output column

    const float* __restrict__ xr = x + (size_t)mrow * D;

    v8f cm = {};
    v8f cv = {};

    #pragma unroll
    for (int kk = 0; kk < 4; ++kk) {
        // A fragment (16x32 f16): lane group 0: K=k0..k0+15
        //                         lane group 1: K=k0+16..k0+31
        const int ka = kk * 32 + hi * 16;
        const float4 a0 = *(const float4*)(xr + ka);
        const float4 a1 = *(const float4*)(xr + ka + 4);
        const float4 a2 = *(const float4*)(xr + ka + 8);
        const float4 a3 = *(const float4*)(xr + ka + 12);
        v16h a;
        a[0]  = (_Float16)a0.x; a[1]  = (_Float16)a0.y;
        a[2]  = (_Float16)a0.z; a[3]  = (_Float16)a0.w;
        a[4]  = (_Float16)a1.x; a[5]  = (_Float16)a1.y;
        a[6]  = (_Float16)a1.z; a[7]  = (_Float16)a1.w;
        a[8]  = (_Float16)a2.x; a[9]  = (_Float16)a2.y;
        a[10] = (_Float16)a2.z; a[11] = (_Float16)a2.w;
        a[12] = (_Float16)a3.x; a[13] = (_Float16)a3.y;
        a[14] = (_Float16)a3.z; a[15] = (_Float16)a3.w;

        // B fragments: contiguous 32 bytes per lane (pre-fragmented layout)
        const size_t foff = ((size_t)(wave * 4 + kk) * 32 + lane) * 16;
        const v16h bmf = *(const v16h*)(Wm + foff);
        const v16h blf = *(const v16h*)(Wl + foff);

        cm = __builtin_amdgcn_wmma_f32_16x16x32_f16(false, a, false, bmf,
                                                    (short)0, cm, false, false);
        cv = __builtin_amdgcn_wmma_f32_16x16x32_f16(false, a, false, blf,
                                                    (short)0, cv, false, false);
    }

    // C/D layout: VGPR r -> M = mtile*16 + r + 8*hi, N = ncol
    const float bmv   = bm[ncol];
    const float blv   = bl[ncol];
    const int   mbase = mtile * 16 + 8 * hi;
    #pragma unroll
    for (int r = 0; r < 8; ++r) {
        const size_t off = (size_t)(mbase + r) * D + ncol;
        h_mean[off] = cm[r] + bmv;
        h_var[off]  = __expf(cv[r] + blv);
    }
}

// ---------------------------------------------------------------------------
// Edge scatter: one thread = (edge, 4-feature group). All gathered/scattered
// arrays are L2-resident (each ~20 MB << 192 MB L2); f32 atomic adds resolve
// in the L2 atomic units via unsafeAtomicAdd -> global_atomic_add_f32.
// ---------------------------------------------------------------------------
__global__ void __launch_bounds__(256) edge_kernel(const long long* __restrict__ ei,
                                                   const float* __restrict__ ewm,
                                                   const float* __restrict__ ewv,
                                                   const float* __restrict__ h_mean,
                                                   const float* __restrict__ h_var,
                                                   float* __restrict__ sum_mean,
                                                   float* __restrict__ sum_var,
                                                   float* __restrict__ degree) {
    const int idx = blockIdx.x * blockDim.x + threadIdx.x;
    const int e   = idx >> 5;
    const int fg  = (idx & 31) << 2;                 // feature base (0,4,..,124)
    if (e >= N_EDGES) return;

    const int   row = (int)ei[e];
    const int   col = (int)ei[N_EDGES + e];
    const float wm  = ewm[e];
    const float wv  = ewv[e];
    const float wm2 = wm * wm;

    const float4 hm = *(const float4*)(h_mean + (size_t)col * D + fg);
    const float4 hv = *(const float4*)(h_var  + (size_t)col * D + fg);

    float4 mmsg, vmsg;
    mmsg.x = hm.x * wm;  vmsg.x = wm2 * hv.x + hm.x * hm.x * wv;
    mmsg.y = hm.y * wm;  vmsg.y = wm2 * hv.y + hm.y * hm.y * wv;
    mmsg.z = hm.z * wm;  vmsg.z = wm2 * hv.z + hm.z * hm.z * wv;
    mmsg.w = hm.w * wm;  vmsg.w = wm2 * hv.w + hm.w * hm.w * wv;

    float* sm = sum_mean + (size_t)row * D + fg;
    float* sv = sum_var  + (size_t)row * D + fg;
    unsafeAtomicAdd(sm + 0, mmsg.x);
    unsafeAtomicAdd(sm + 1, mmsg.y);
    unsafeAtomicAdd(sm + 2, mmsg.z);
    unsafeAtomicAdd(sm + 3, mmsg.w);
    unsafeAtomicAdd(sv + 0, vmsg.x);
    unsafeAtomicAdd(sv + 1, vmsg.y);
    unsafeAtomicAdd(sv + 2, vmsg.z);
    unsafeAtomicAdd(sv + 3, vmsg.w);

    if (fg == 0) unsafeAtomicAdd(degree + row, 1.0f);
}

// ---------------------------------------------------------------------------
// Finalize: degree normalization + LayerNorm(out_mean) in place in d_out.
// One 128-thread block per node; LDS tree reduction for mean / variance.
// ---------------------------------------------------------------------------
__global__ void __launch_bounds__(128) finalize_kernel(float* __restrict__ out,
                                                       const float* __restrict__ degree,
                                                       const float* __restrict__ gamma,
                                                       const float* __restrict__ beta) {
    __shared__ float red[128];
    const int node = blockIdx.x;
    const int f    = threadIdx.x;

    float deg = degree[node];
    deg = deg < 1.0f ? 1.0f : deg;

    const size_t off  = (size_t)node * D + f;
    const size_t voff = (size_t)N_NODES * D + off;

    const float om = out[off]  / deg;
    const float ov = out[voff] / (deg * deg);

    // mean
    red[f] = om;
    __syncthreads();
    #pragma unroll
    for (int s = 64; s > 0; s >>= 1) {
        if (f < s) red[f] += red[f + s];
        __syncthreads();
    }
    const float mu = red[0] * (1.0f / D);
    __syncthreads();

    // variance
    const float d = om - mu;
    red[f] = d * d;
    __syncthreads();
    #pragma unroll
    for (int s = 64; s > 0; s >>= 1) {
        if (f < s) red[f] += red[f + s];
        __syncthreads();
    }
    const float var = red[0] * (1.0f / D);
    const float inv = rsqrtf(var + LN_EPS);

    out[off]  = d * inv * gamma[f] + beta[f];
    out[voff] = ov;
}

// ---------------------------------------------------------------------------
extern "C" void kernel_launch(void* const* d_in, const int* in_sizes, int n_in,
                              void* d_out, int out_size, void* d_ws, size_t ws_size,
                              hipStream_t stream) {
    const float*     x        = (const float*)d_in[0];
    const long long* ei       = (const long long*)d_in[1];   // int64 per reference
    const float*     ewm      = (const float*)d_in[2];
    const float*     ewv      = (const float*)d_in[3];
    const float*     W_mean   = (const float*)d_in[4];
    const float*     b_mean   = (const float*)d_in[5];
    const float*     W_logvar = (const float*)d_in[6];
    const float*     b_logvar = (const float*)d_in[7];
    const float*     gamma    = (const float*)d_in[8];
    const float*     beta     = (const float*)d_in[9];

    float* out = (float*)d_out;   // [out_mean | out_var], also used as accumulators

    // Workspace layout
    float*    h_mean = (float*)d_ws;
    float*    h_var  = h_mean + (size_t)N_NODES * D;
    float*    degree = h_var  + (size_t)N_NODES * D;
    _Float16* Wm16   = (_Float16*)(degree + N_NODES);        // fragment-ordered
    _Float16* Wl16   = Wm16 + D * D;

    // 1. zero accumulators (d_out) + degree
    {
        const size_t total = (size_t)2 * N_NODES * D + N_NODES;
        const int blocks = (int)((total + 255) / 256);
        zero_kernel<<<blocks, 256, 0, stream>>>(out, degree);
    }
    // 2. convert weights to f16 in B-fragment order
    convw_frag_kernel<<<4, 256, 0, stream>>>(W_mean, W_logvar, Wm16, Wl16);

    // 3. WMMA GEMMs -> h_mean, h_var
    gemm_kernel<<<N_NODES / 16, 256, 0, stream>>>(x, Wm16, Wl16, b_mean, b_logvar,
                                                  h_mean, h_var);

    // 4. edge messages + segment-sum scatter (accumulate into d_out)
    {
        const long long threads = (long long)N_EDGES * 32;
        const int blocks = (int)((threads + 255) / 256);
        edge_kernel<<<blocks, 256, 0, stream>>>(ei, ewm, ewv, h_mean, h_var,
                                                out, out + (size_t)N_NODES * D, degree);
    }

    // 5. normalize + LayerNorm in place
    finalize_kernel<<<N_NODES, 128, 0, stream>>>(out, degree, gamma, beta);
}
